// PiFoldModel_1279900254907
// MI455X (gfx1250) — matrix operand order repaired and checked by reading, not compiled, not run on hardware
//
#include <hip/hip_runtime.h>
#include <math.h>

typedef _Float16 half_t;
typedef __attribute__((ext_vector_type(16))) _Float16 v16h;
typedef __attribute__((ext_vector_type(8)))  float    v8f;

#define NN 4096              // nodes
#define NE 122880            // edges
#define NG 16                // graphs
#define HDIM 128

static inline int ceil32i(int x){ return (x + 31) & ~31; }
static inline int ceil64i(int x){ return (x + 63) & ~63; }

__device__ __forceinline__ float actf(float v, int act){
  switch(act){
    case 1: return fmaxf(v, 0.f);                                     // relu
    case 2: return 0.5f * v * (1.f + erff(v * 0.70710678118654752f)); // gelu (exact)
    case 3: return v > 0.f ? v : 0.01f * v;                           // leaky 0.01
    case 4: return 1.f / (1.f + expf(-v));                            // sigmoid
    default: return v;
  }
}

// ---------------------------------------------------------------------------
// WMMA GEMM: C[M,N] = act(A[M,K] @ W[K,N] + bias)
// A: f16 row-major, row stride lda (>= K, K multiple of 32, zero padded)
// Wt: f16 W transposed, [ceil64(N) rows x ldw cols], ldw = K
// One wave per block computes a 16x64 output tile. All 5 fragment loads
// (A + B0..B3) are issued up front each k-step, then 4 independent WMMAs are
// issued back-to-back: loads overlap WMMA issue and the WMMA->VALU hazard
// NOPs fire at most once per iteration. grid = (M/16, ceil64(N)/64), block=32.
// ---------------------------------------------------------------------------
__global__ void k_gemm(const half_t* __restrict__ A, int lda,
                       const half_t* __restrict__ Wt, int ldw,
                       const float* __restrict__ bias,
                       float* __restrict__ Cf, half_t* __restrict__ Ch,
                       int ldc, int K, int N, int act)
{
  const int lane = threadIdx.x;
  const int lid  = lane & 15;
  const int sel  = lane >> 4;              // 0: lanes 0-15, 1: lanes 16-31
  const size_t m0 = (size_t)blockIdx.x * 16;
  const int nb = blockIdx.y * 64;
  const half_t* ar  = A + (m0 + (size_t)lid) * (size_t)lda;
  const half_t* br0 = Wt + (size_t)(nb + lid) * (size_t)ldw;
  const half_t* br1 = br0 + (size_t)16 * (size_t)ldw;
  const half_t* br2 = br0 + (size_t)32 * (size_t)ldw;
  const half_t* br3 = br0 + (size_t)48 * (size_t)ldw;
  v8f acc0 = {}, acc1 = {}, acc2 = {}, acc3 = {};
  for (int k = 0; k < K; k += 32) {
    union U { uint4 q[2]; v16h h; } a, b0, b1, b2, b3;
    // A 16x32 f16 fragment: lanes 0-15 K 0..15, lanes 16-31 K 16..31
    a.q[0]  = *(const uint4*)(ar  + k + sel * 16);
    a.q[1]  = *(const uint4*)(ar  + k + sel * 16 + 8);
    // B 32x16 f16 fragments: lanes 0-15 K 0..15, lanes 16-31 K 16..31
    b0.q[0] = *(const uint4*)(br0 + k + sel * 16);
    b0.q[1] = *(const uint4*)(br0 + k + sel * 16 + 8);
    b1.q[0] = *(const uint4*)(br1 + k + sel * 16);
    b1.q[1] = *(const uint4*)(br1 + k + sel * 16 + 8);
    b2.q[0] = *(const uint4*)(br2 + k + sel * 16);
    b2.q[1] = *(const uint4*)(br2 + k + sel * 16 + 8);
    b3.q[0] = *(const uint4*)(br3 + k + sel * 16);
    b3.q[1] = *(const uint4*)(br3 + k + sel * 16 + 8);
    acc0 = __builtin_amdgcn_wmma_f32_16x16x32_f16(false, a.h, false, b0.h,
                                                  (short)0, acc0, false, false);
    acc1 = __builtin_amdgcn_wmma_f32_16x16x32_f16(false, a.h, false, b1.h,
                                                  (short)0, acc1, false, false);
    acc2 = __builtin_amdgcn_wmma_f32_16x16x32_f16(false, a.h, false, b2.h,
                                                  (short)0, acc2, false, false);
    acc3 = __builtin_amdgcn_wmma_f32_16x16x32_f16(false, a.h, false, b3.h,
                                                  (short)0, acc3, false, false);
  }
  v8f accs[4] = {acc0, acc1, acc2, acc3};
#pragma unroll
  for (int t = 0; t < 4; ++t) {
    const int n = nb + t * 16 + lid;
    if (n >= N) continue;
    const float bv = bias ? bias[n] : 0.f;
#pragma unroll
    for (int r = 0; r < 8; ++r) {
      const size_t m = m0 + r + sel * 8;   // D: VGPR r -> row r (+8 for hi lanes)
      const float v = actf(accs[t][r] + bv, act);
      if (Cf) Cf[m * (size_t)ldc + n] = v;
      if (Ch) Ch[m * (size_t)ldc + n] = (half_t)v;
    }
  }
}

// W[k,dout] f32 -> Wt[n,k] f16, zero-padded to [Np x Kp]
__global__ void k_wt16(const float* __restrict__ W, half_t* __restrict__ Wt,
                       int din, int dout, int Kp, int Np){
  size_t i = (size_t)blockIdx.x * 256 + threadIdx.x;
  size_t tot = (size_t)Np * Kp;
  if (i >= tot) return;
  int n = (int)(i / Kp), k = (int)(i % Kp);
  float v = (n < dout && k < din) ? W[(size_t)k * dout + n] : 0.f;
  Wt[i] = (half_t)v;
}

// f32 [M,C] -> f16 [M,Cp] zero padded
__global__ void k_a16(const float* __restrict__ X, half_t* __restrict__ Y,
                      long M, int C, int Cp){
  size_t i = (size_t)blockIdx.x * 256 + threadIdx.x;
  if (i >= (size_t)M * Cp) return;
  size_t m = i / Cp; int c = (int)(i % Cp);
  Y[i] = (half_t)(c < C ? X[m * (size_t)C + c] : 0.f);
}

// [hV[src] | hE | hV[dst]] -> f16 [E,384]
__global__ void k_concat3(const float* __restrict__ hV, const float* __restrict__ hE,
                          const int* __restrict__ src, const int* __restrict__ dst,
                          half_t* __restrict__ out){
  size_t i = (size_t)blockIdx.x * 256 + threadIdx.x;
  if (i >= (size_t)NE * 384) return;
  size_t e = i / 384; int c = (int)(i % 384);
  float v;
  if (c < 128)       v = hV[(size_t)src[e] * HDIM + c];
  else if (c < 256)  v = hE[e * HDIM + (c - 128)];
  else               v = hV[(size_t)dst[e] * HDIM + (c - 256)];
  out[i] = (half_t)v;
}

// column mean / rsqrt(var+eps) over M rows of (A [+ B])
__global__ void k_bnstats(const float* __restrict__ A, const float* __restrict__ B,
                          float* __restrict__ mu, float* __restrict__ rs, long M){
  __shared__ float s1[256], s2[256];
  const int c = blockIdx.x, t = threadIdx.x;
  float a = 0.f, b = 0.f;
  for (long m = t; m < M; m += 256){
    float v = A[(size_t)m * HDIM + c];
    if (B) v += B[(size_t)m * HDIM + c];
    a += v; b += v * v;
  }
  s1[t] = a; s2[t] = b; __syncthreads();
  for (int o = 128; o > 0; o >>= 1){
    if (t < o){ s1[t] += s1[t + o]; s2[t] += s2[t + o]; }
    __syncthreads();
  }
  if (t == 0){
    float mean = s1[0] / (float)M;
    float var  = s2[0] / (float)M - mean * mean;
    mu[c] = mean; rs[c] = rsqrtf(fmaxf(var, 0.f) + 1e-5f);
  }
}

__global__ void k_bnapply(const float* __restrict__ A, const float* __restrict__ B,
                          const float* __restrict__ mu, const float* __restrict__ rs,
                          const float* __restrict__ g, const float* __restrict__ bb,
                          float* __restrict__ O, long M){
  size_t i = (size_t)blockIdx.x * 256 + threadIdx.x;
  if (i >= (size_t)M * HDIM) return;
  int c = (int)(i % HDIM);
  float v = A[i]; if (B) v += B[i];
  O[i] = (v - mu[c]) * rs[c] * g[c] + bb[c];
}

__global__ void k_fill(float* __restrict__ p, float v, size_t n){
  size_t i = (size_t)blockIdx.x * 256 + threadIdx.x;
  if (i < n) p[i] = v;
}

__device__ __forceinline__ void atomicMaxF(float* addr, float val){
  unsigned int* a = (unsigned int*)addr;
  unsigned int old = *a;
  while (true){
    float f = __uint_as_float(old);
    if (f >= val) break;
    unsigned int assumed = old;
    old = atomicCAS(a, assumed, __float_as_uint(val));
    if (old == assumed) break;
  }
}

__global__ void k_segmax(const float* __restrict__ lg, const int* __restrict__ src,
                         float* __restrict__ mx, float scale){
  size_t i = (size_t)blockIdx.x * 256 + threadIdx.x;
  if (i >= (size_t)NE * 4) return;
  size_t e = i / 4; int h = (int)(i % 4);
  atomicMaxF(&mx[(size_t)src[e] * 4 + h], lg[i] * scale);
}

__global__ void k_segexp(const float* __restrict__ lg, const int* __restrict__ src,
                         const float* __restrict__ mx, float* __restrict__ ex,
                         float* __restrict__ sm, float scale){
  size_t i = (size_t)blockIdx.x * 256 + threadIdx.x;
  if (i >= (size_t)NE * 4) return;
  size_t e = i / 4; int h = (int)(i % 4);
  float v = expf(lg[i] * scale - mx[(size_t)src[e] * 4 + h]);
  ex[i] = v;
  atomicAdd(&sm[(size_t)src[e] * 4 + h], v);
}

__global__ void k_segdiv(float* __restrict__ ex, const int* __restrict__ src,
                         const float* __restrict__ sm){
  size_t i = (size_t)blockIdx.x * 256 + threadIdx.x;
  if (i >= (size_t)NE * 4) return;
  size_t e = i / 4; int h = (int)(i % 4);
  ex[i] /= sm[(size_t)src[e] * 4 + h];
}

// agg[src[e], c] += att[e, c/32] * V[e, c]
__global__ void k_agg(const float* __restrict__ att, const float* __restrict__ V,
                      const int* __restrict__ src, float* __restrict__ agg){
  size_t i = (size_t)blockIdx.x * 256 + threadIdx.x;
  if (i >= (size_t)NE * HDIM) return;
  size_t e = i / HDIM; int c = (int)(i % HDIM);
  atomicAdd(&agg[(size_t)src[e] * HDIM + c], att[e * 4 + (c >> 5)] * V[i]);
}

__global__ void k_cnt(const int* __restrict__ bid, float* __restrict__ cnt){
  int i = blockIdx.x * 256 + threadIdx.x;
  if (i < NN) atomicAdd(&cnt[bid[i]], 1.f);
}

__global__ void k_ctxsum(const float* __restrict__ hV, const int* __restrict__ bid,
                         float* __restrict__ cV){
  size_t i = (size_t)blockIdx.x * 256 + threadIdx.x;
  if (i >= (size_t)NN * HDIM) return;
  size_t m = i / HDIM; int c = (int)(i % HDIM);
  atomicAdd(&cV[(size_t)bid[m] * HDIM + c], hV[i]);
}

__global__ void k_ctxdiv(const float* __restrict__ cV, const float* __restrict__ cnt,
                         half_t* __restrict__ o16){
  int i = blockIdx.x * 256 + threadIdx.x;
  if (i >= NG * HDIM) return;
  o16[i] = (half_t)(cV[i] / fmaxf(cnt[i / HDIM], 1.f));
}

__global__ void k_gate(float* __restrict__ hV, const float* __restrict__ gate,
                       const int* __restrict__ bid){
  size_t i = (size_t)blockIdx.x * 256 + threadIdx.x;
  if (i >= (size_t)NN * HDIM) return;
  size_t m = i / HDIM; int c = (int)(i % HDIM);
  hV[i] *= gate[(size_t)bid[m] * HDIM + c];
}

__global__ void k_lsm(const float* __restrict__ X, float* __restrict__ O, int C){
  int m = blockIdx.x * 256 + threadIdx.x;
  if (m >= NN) return;
  const float* x = X + (size_t)m * C;
  float mx = -1e30f;
  for (int c = 0; c < C; ++c) mx = fmaxf(mx, x[c]);
  float s = 0.f;
  for (int c = 0; c < C; ++c) s += expf(x[c] - mx);
  float ls = logf(s) + mx;
  for (int c = 0; c < C; ++c) O[(size_t)m * C + c] = x[c] - ls;
}

// ---------------------------------------------------------------------------
extern "C" void kernel_launch(void* const* d_in, const int* in_sizes, int n_in,
                              void* d_out, int out_size, void* d_ws, size_t ws_size,
                              hipStream_t stream)
{
  (void)in_sizes; (void)out_size;
  if (n_in < 376) return;

  const float* hVin = (const float*)d_in[0];          // [4096,213]
  const float* hPin = (const float*)d_in[1];          // [122880,416]
  const int*   src  = (const int*)d_in[2];            // [2,E]
  const int*   dst  = src + NE;
  const int*   bid  = (const int*)d_in[3];            // [4096]
  auto par = [&](int i)->const float*{ return (const float*)d_in[i]; };

  // params flattened jax-style (sorted keys, depth-first), offset 4
  const int iWe_b=4, iWe_w=5;
  const int iWv_bn1_b=6, iWv_bn1_g=7, iWv_bn2_b=8, iWv_bn2_g=9;
  const int iWv_l1_b=10, iWv_l1_w=11, iWv_l2_b=12, iWv_l2_w=13, iWv_l3_b=14, iWv_l3_w=15;
  const int iEemb_b=16, iEemb_w=17;
  const int iLay=18, LS=35;
  // per-layer: 0 B1.b 1 B1.w 2 B2.b 3 B2.w 4 B3.b 5 B3.w 6 WO.w 7 WV1.b 8 WV1.w
  // 9 WV2.b 10 WV2.w 11 WV3.b 12 WV3.w 13 bn0.b 14 bn0.g 15 bn1.b 16 bn1.g
  // 17 g1.b 18 g1.w 19 g2.b 20 g2.w 21 g3.b 22 g3.w 23 dl1.b 24 dl1.w
  // 25 dl2.b 26 dl2.w 27 W11.b 28 W11.w 29 W12.b 30 W12.w 31 W13.b 32 W13.w
  // 33 ebn.b 34 ebn.g
  const int iNemb_b=368, iNemb_w=369, iNEd_b=370, iNEd_g=371, iNNo_b=372, iNNo_g=373;
  const int iRO_b=374, iRO_w=375;

  // ---- workspace bump allocator ----
  char* base = (char*)d_ws; size_t off = 0;
  auto alloc = [&](size_t bytes)->void*{
    void* p = base + off; off = (off + bytes + 255) & ~(size_t)255; return p;
  };

  auto prep = [&](const float* W, int din, int dout)->half_t*{
    int Kp = ceil32i(din), Np = ceil64i(dout);
    half_t* p = (half_t*)alloc((size_t)Kp * Np * sizeof(half_t));
    size_t tot = (size_t)Kp * Np;
    k_wt16<<<dim3((unsigned)((tot + 255) / 256)), 256, 0, stream>>>(W, p, din, dout, Kp, Np);
    return p;
  };

  auto gemm = [&](const half_t* A, int lda, const half_t* Wt, int Kp, int dout,
                  const float* bias, float* Cf, half_t* Ch, int ldc, long M, int act){
    dim3 g((unsigned)(M / 16), (unsigned)(ceil64i(dout) / 64));
    k_gemm<<<g, 32, 0, stream>>>(A, lda, Wt, Kp, bias, Cf, Ch, ldc, Kp, dout, act);
  };

  // ---- weights -> f16 transposed ----
  half_t* wNemb = prep(par(iNemb_w), 213, 128);
  half_t* wEemb = prep(par(iEemb_w), 416, 128);
  half_t* wWe   = prep(par(iWe_w),   128, 128);
  half_t* wL1   = prep(par(iWv_l1_w),128, 128);
  half_t* wL2   = prep(par(iWv_l2_w),128, 128);
  half_t* wL3   = prep(par(iWv_l3_w),128, 128);
  half_t* wRO   = prep(par(iRO_w),   128, 20);
  half_t *wB1[10], *wB2[10], *wB3[10], *wWO[10], *wV1[10], *wV2[10], *wV3[10];
  half_t *wG1[10], *wG2[10], *wG3[10], *wD1[10], *wD2[10], *w11[10], *w12[10], *w13[10];
  for (int L = 0; L < 10; ++L){
    int b = iLay + LS * L;
    wB1[L] = prep(par(b + 1),  384, 128);
    wB2[L] = prep(par(b + 3),  128, 128);
    wB3[L] = prep(par(b + 5),  128, 4);
    wWO[L] = prep(par(b + 6),  128, 128);
    wV1[L] = prep(par(b + 8),  256, 128);
    wV2[L] = prep(par(b + 10), 128, 128);
    wV3[L] = prep(par(b + 12), 128, 128);
    wG1[L] = prep(par(b + 18), 128, 128);
    wG2[L] = prep(par(b + 20), 128, 128);
    wG3[L] = prep(par(b + 22), 128, 128);
    wD1[L] = prep(par(b + 24), 128, 512);
    wD2[L] = prep(par(b + 26), 512, 128);
    w11[L] = prep(par(b + 28), 384, 128);
    w12[L] = prep(par(b + 30), 128, 128);
    w13[L] = prep(par(b + 32), 128, 128);
  }

  // ---- activation buffers ----
  half_t* nh16 = (half_t*)alloc((size_t)NN * 224 * 2);   // node input f16 (K pad 224)
  half_t* eX3  = (half_t*)alloc((size_t)NE * 416 * 2);   // edge input f16 / concat [E,384]
  float*  hV   = (float*) alloc((size_t)NN * HDIM * 4);
  float*  hE   = (float*) alloc((size_t)NE * HDIM * 4);
  float*  tE1  = (float*) alloc((size_t)NE * HDIM * 4);  // V / msg
  half_t* e16a = (half_t*)alloc((size_t)NE * HDIM * 2);
  half_t* e16b = (half_t*)alloc((size_t)NE * HDIM * 2);
  float*  lgE  = (float*) alloc((size_t)NE * 4 * 4);
  float*  exE  = (float*) alloc((size_t)NE * 4 * 4);
  float*  mx   = (float*) alloc((size_t)NN * 4 * 4);
  float*  sm   = (float*) alloc((size_t)NN * 4 * 4);
  float*  agg  = (float*) alloc((size_t)NN * HDIM * 4);
  float*  nf1  = (float*) alloc((size_t)NN * HDIM * 4);
  half_t* n16b = (half_t*)alloc((size_t)NN * HDIM * 2);
  half_t* n16a = (half_t*)alloc((size_t)NN * 512 * 2);
  float*  cV   = (float*) alloc((size_t)NG * HDIM * 4);
  half_t* cv16 = (half_t*)alloc((size_t)NG * HDIM * 2);
  half_t* g16a = (half_t*)alloc((size_t)NG * HDIM * 2);
  half_t* g16b = (half_t*)alloc((size_t)NG * HDIM * 2);
  float*  gate = (float*) alloc((size_t)NG * HDIM * 4);
  float*  cnt  = (float*) alloc(NG * 4);
  float*  mu   = (float*) alloc(HDIM * 4);
  float*  rs   = (float*) alloc(HDIM * 4);
  float*  rlog = (float*) alloc((size_t)NN * 32 * 4);
  if (off > ws_size) return;   // workspace too small: bail

  auto blocks = [](size_t n){ return dim3((unsigned)((n + 255) / 256)); };
  auto bn = [&](const float* A, const float* B, const float* g, const float* b,
                float* O, long M){
    k_bnstats<<<HDIM, 256, 0, stream>>>(A, B, mu, rs, M);
    k_bnapply<<<blocks((size_t)M * HDIM), 256, 0, stream>>>(A, B, mu, rs, g, b, O, M);
  };
  const float scale = 0.17677669529663688f;   // 1/sqrt(32)

  // ---- encoders ----
  k_a16<<<blocks((size_t)NN * 224), 256, 0, stream>>>(hVin, nh16, NN, 213, 224);
  gemm(nh16, 224, wNemb, 224, 128, par(iNemb_b), nf1, nullptr, 128, NN, 0);
  bn(nf1, nullptr, par(iNNo_g), par(iNNo_b), nf1, NN);
  k_a16<<<blocks((size_t)NN * 128), 256, 0, stream>>>(nf1, n16b, NN, 128, 128);
  gemm(n16b, 128, wL1, 128, 128, par(iWv_l1_b), nf1, nullptr, 128, NN, 3);
  bn(nf1, nullptr, par(iWv_bn1_g), par(iWv_bn1_b), nf1, NN);
  k_a16<<<blocks((size_t)NN * 128), 256, 0, stream>>>(nf1, n16b, NN, 128, 128);
  gemm(n16b, 128, wL2, 128, 128, par(iWv_l2_b), nf1, nullptr, 128, NN, 3);
  bn(nf1, nullptr, par(iWv_bn2_g), par(iWv_bn2_b), nf1, NN);
  k_a16<<<blocks((size_t)NN * 128), 256, 0, stream>>>(nf1, n16b, NN, 128, 128);
  gemm(n16b, 128, wL3, 128, 128, par(iWv_l3_b), hV, nullptr, 128, NN, 0);

  k_a16<<<blocks((size_t)NE * 416), 256, 0, stream>>>(hPin, eX3, NE, 416, 416);
  gemm(eX3, 416, wEemb, 416, 128, par(iEemb_b), tE1, nullptr, 128, NE, 0);
  bn(tE1, nullptr, par(iNEd_g), par(iNEd_b), tE1, NE);
  k_a16<<<blocks((size_t)NE * 128), 256, 0, stream>>>(tE1, e16a, NE, 128, 128);
  gemm(e16a, 128, wWe, 128, 128, par(iWe_b), hE, nullptr, 128, NE, 0);

  // ---- GNN layers ----
  for (int L = 0; L < 10; ++L){
    int b = iLay + LS * L;
    // attention logits
    k_concat3<<<blocks((size_t)NE * 384), 256, 0, stream>>>(hV, hE, src, dst, eX3);
    gemm(eX3, 384, wB1[L], 384, 128, par(b + 0), nullptr, e16a, 128, NE, 1);
    gemm(e16a, 128, wB2[L], 128, 128, par(b + 2), nullptr, e16b, 128, NE, 1);
    gemm(e16b, 128, wB3[L], 128, 4, par(b + 4), lgE, nullptr, 4, NE, 0);
    // segment softmax over src
    k_fill<<<blocks((size_t)NN * 4), 256, 0, stream>>>(mx, -1e30f, (size_t)NN * 4);
    k_segmax<<<blocks((size_t)NE * 4), 256, 0, stream>>>(lgE, src, mx, scale);
    k_fill<<<blocks((size_t)NN * 4), 256, 0, stream>>>(sm, 0.f, (size_t)NN * 4);
    k_segexp<<<blocks((size_t)NE * 4), 256, 0, stream>>>(lgE, src, mx, exE, sm, scale);
    k_segdiv<<<blocks((size_t)NE * 4), 256, 0, stream>>>(exE, src, sm);
    // V path (reads [hE | hV[dst]] = columns 128..383 of eX3)
    gemm(eX3 + 128, 384, wV1[L], 256, 128, par(b + 7), nullptr, e16a, 128, NE, 2);
    gemm(e16a, 128, wV2[L], 128, 128, par(b + 9), nullptr, e16b, 128, NE, 2);
    gemm(e16b, 128, wV3[L], 128, 128, par(b + 11), tE1, nullptr, 128, NE, 0);
    // weighted aggregation
    k_fill<<<blocks((size_t)NN * HDIM), 256, 0, stream>>>(agg, 0.f, (size_t)NN * HDIM);
    k_agg<<<blocks((size_t)NE * HDIM), 256, 0, stream>>>(exE, tE1, src, agg);
    // h_V = bn0(h_V + WO @ agg)
    k_a16<<<blocks((size_t)NN * 128), 256, 0, stream>>>(agg, n16b, NN, 128, 128);
    gemm(n16b, 128, wWO[L], 128, 128, nullptr, nf1, nullptr, 128, NN, 0);
    bn(hV, nf1, par(b + 14), par(b + 13), hV, NN);
    // h_V = bn1(h_V + l2(relu(l1 h_V)))
    k_a16<<<blocks((size_t)NN * 128), 256, 0, stream>>>(hV, n16b, NN, 128, 128);
    gemm(n16b, 128, wD1[L], 128, 512, par(b + 23), nullptr, n16a, 512, NN, 1);
    gemm(n16a, 512, wD2[L], 512, 128, par(b + 25), nf1, nullptr, 128, NN, 0);
    bn(hV, nf1, par(b + 16), par(b + 15), hV, NN);
    // edge update
    k_concat3<<<blocks((size_t)NE * 384), 256, 0, stream>>>(hV, hE, src, dst, eX3);
    gemm(eX3, 384, w11[L], 384, 128, par(b + 27), nullptr, e16a, 128, NE, 2);
    gemm(e16a, 128, w12[L], 128, 128, par(b + 29), nullptr, e16b, 128, NE, 2);
    gemm(e16b, 128, w13[L], 128, 128, par(b + 31), tE1, nullptr, 128, NE, 0);
    bn(hE, tE1, par(b + 34), par(b + 33), hE, NE);
    // node-context gating
    k_fill<<<1, 256, 0, stream>>>(cnt, 0.f, NG);
    k_cnt<<<blocks(NN), 256, 0, stream>>>(bid, cnt);
    k_fill<<<blocks((size_t)NG * HDIM), 256, 0, stream>>>(cV, 0.f, (size_t)NG * HDIM);
    k_ctxsum<<<blocks((size_t)NN * HDIM), 256, 0, stream>>>(hV, bid, cV);
    k_ctxdiv<<<blocks((size_t)NG * HDIM), 256, 0, stream>>>(cV, cnt, cv16);
    gemm(cv16, 128, wG1[L], 128, 128, par(b + 17), nullptr, g16a, 128, NG, 1);
    gemm(g16a, 128, wG2[L], 128, 128, par(b + 19), nullptr, g16b, 128, NG, 1);
    gemm(g16b, 128, wG3[L], 128, 128, par(b + 21), gate, nullptr, 128, NG, 4);
    k_gate<<<blocks((size_t)NN * HDIM), 256, 0, stream>>>(hV, gate, bid);
  }

  // ---- readout + log_softmax ----
  k_a16<<<blocks((size_t)NN * 128), 256, 0, stream>>>(hV, n16b, NN, 128, 128);
  gemm(n16b, 128, wRO, 128, 20, par(iRO_b), rlog, nullptr, 20, NN, 0);
  k_lsm<<<blocks(NN), 256, 0, stream>>>(rlog, (float*)d_out, 20);
}
